// ExternalMemory_82789789598188
// MI455X (gfx1250) — compile-verified, matrix operand order checked by reading, not run on hardware
//
#include <hip/hip_runtime.h>
#include <hip/hip_bf16.h>

// ---------------------------------------------------------------------------
// MI455X (gfx1250) memory-bank cross-attention + LRU scatter update.
// GEMMs: v_wmma_f32_16x16x32_bf16 (wave32, 16x16 C tiles, f32 accumulate).
// kvproj uses the Tensor Data Mover: tensor_load_to_lds stages the memory
// tile, tensor_store_from_lds performs the new_memory passthrough copy.
// ---------------------------------------------------------------------------

typedef __attribute__((ext_vector_type(16))) __bf16 v16bf;
typedef __attribute__((ext_vector_type(8)))  float  v8f;
typedef __attribute__((ext_vector_type(4)))  unsigned int u32x4;
typedef __attribute__((ext_vector_type(4)))  int    i32x4;
typedef __attribute__((ext_vector_type(8)))  int    i32x8;

#define WMMA_BF16(A, B, C) \
  __builtin_amdgcn_wmma_f32_16x16x32_bf16(false, (A), false, (B), (short)0, (C), false, false)

#define NB   128      // batch
#define NM   65536    // memory slots
#define ND   512      // model dim
#define NH   8        // heads
#define HD   64       // head dim
#define SPLITS 64     // flash split-K factor

#if __has_builtin(__builtin_amdgcn_tensor_load_to_lds) && \
    __has_builtin(__builtin_amdgcn_tensor_store_from_lds) && \
    __has_builtin(__builtin_amdgcn_s_wait_tensorcnt)
#define USE_TDM 1
#else
#define USE_TDM 0
#endif

// Load 16 contiguous bf16 (32B) as a WMMA half-K fragment. 16B-aligned.
__device__ __forceinline__ v16bf load_frag(const __bf16* p) {
  union { v16bf v; uint4 q[2]; } u;
  u.q[0] = *reinterpret_cast<const uint4*>(p);
  u.q[1] = *reinterpret_cast<const uint4*>(p + 8);
  return u.v;
}

#if USE_TDM
// D# group 0: count=1 descriptor, LDS byte address, 57-bit global address,
// type=2 ("image") in bits [127:126].
__device__ __forceinline__ u32x4 tdm_g0(unsigned long long ga, unsigned lds) {
  u32x4 g0;
  g0[0] = 1u;                                              // count = 1, user mode
  g0[1] = lds;                                             // lds_addr (bytes)
  g0[2] = (unsigned)(ga & 0xFFFFFFFFull);                  // global_addr[31:0]
  g0[3] = ((unsigned)(ga >> 32) & 0x01FFFFFFu) | 0x80000000u;  // [56:32] | type=2
  return g0;
}
// D# group 1: 2D fp32 tile rows x 512, tensor_dim0_stride = 512 elements.
__device__ __forceinline__ i32x8 tdm_g1(int rows) {
  i32x8 g1;
  g1[0] = (2 << 16);        // wg_mask=0, data_size=2 (4 bytes)
  g1[1] = (512 << 16);      // tensor_dim0[15:0] in bits[63:48]
  g1[2] = (rows << 16);     // tensor_dim1[15:0] in bits[95:80]
  g1[3] = (512 << 16);      // tile_dim0 = 512 in bits[127:112]
  g1[4] = rows;             // tile_dim1 in bits[143:128]; tile_dim2 = 0
  g1[5] = 512;              // tensor_dim0_stride[31:0] in bits[191:160]
  g1[6] = 0;
  g1[7] = 0;
  return g1;
}

__device__ __forceinline__ void tdm_load_tile(unsigned long long ga, unsigned lds, int rows) {
  i32x4 z4 = {0, 0, 0, 0};
#if __clang_major__ >= 23
  i32x8 z8 = {0, 0, 0, 0, 0, 0, 0, 0};
  __builtin_amdgcn_tensor_load_to_lds(tdm_g0(ga, lds), tdm_g1(rows), z4, z4, z8, 0);
#else
  __builtin_amdgcn_tensor_load_to_lds(tdm_g0(ga, lds), tdm_g1(rows), z4, z4, 0);
#endif
}

__device__ __forceinline__ void tdm_store_tile(unsigned long long ga, unsigned lds, int rows) {
  i32x4 z4 = {0, 0, 0, 0};
#if __clang_major__ >= 23
  i32x8 z8 = {0, 0, 0, 0, 0, 0, 0, 0};
  __builtin_amdgcn_tensor_store_from_lds(tdm_g0(ga, lds), tdm_g1(rows), z4, z4, z8, 0);
#else
  __builtin_amdgcn_tensor_store_from_lds(tdm_g0(ga, lds), tdm_g1(rows), z4, z4, 0);
#endif
}
#endif  // USE_TDM

// ---------------------------------------------------------------------------
// prep: f32->bf16 weight/feature conversion, usage passthrough, hist zero
// ---------------------------------------------------------------------------
__global__ void prep_kernel(const float* Wq, const float* Wk, const float* Wv,
                            const float* Wo, const float* feat, const float* usage,
                            __bf16* wq_b, __bf16* wk_b, __bf16* wv_b, __bf16* wo_b,
                            __bf16* feat_b, float* usage_out, unsigned* hist) {
  int i = blockIdx.x * 256 + threadIdx.x;           // grid covers 512*512
  wq_b[i] = (__bf16)Wq[i];
  wk_b[i] = (__bf16)Wk[i];
  wv_b[i] = (__bf16)Wv[i];
  wo_b[i] = (__bf16)Wo[i];
  if (i < NB * ND)  feat_b[i]    = (__bf16)feat[i];
  if (i < NM)       usage_out[i] = usage[i];
  if (i < 4096)     hist[i] = 0u;
}

// ---------------------------------------------------------------------------
// qproj: q = (feat @ Wq^T + bq) * (1/sqrt(hd)), bf16 out [NB, ND]
// ---------------------------------------------------------------------------
__global__ __launch_bounds__(256) void qproj_kernel(const __bf16* feat_b,
                                                    const __bf16* wq_b,
                                                    const float* bq, __bf16* q_b) {
  int lane = threadIdx.x & 31, wave = threadIdx.x >> 5;
  int tile = blockIdx.x * 8 + wave;
  int bt = tile >> 5, dt = tile & 31;
  int col = lane & 15, seg = lane >> 4;
  float bias = bq[dt * 16 + col];
  v8f acc;
  #pragma unroll
  for (int r = 0; r < 8; r++) acc[r] = bias;
  #pragma unroll
  for (int ks = 0; ks < 16; ks++) {
    v16bf a = load_frag(feat_b + (bt * 16 + col) * ND + ks * 32 + seg * 16);
    v16bf b = load_frag(wq_b   + (dt * 16 + col) * ND + ks * 32 + seg * 16);
    acc = WMMA_BF16(a, b, acc);
  }
  #pragma unroll
  for (int r = 0; r < 8; r++) {
    int brow = bt * 16 + r + 8 * seg;
    q_b[brow * ND + dt * 16 + col] = (__bf16)(acc[r] * 0.125f);
  }
}

// ---------------------------------------------------------------------------
// kvproj: per 32-row memory block:
//   TDM: global memory tile -> LDS (f32), LDS -> new_memory (async copy)
//   all waves: LDS f32 -> padded LDS bf16, then
//   K  = mem @ Wk^T + bk    -> bf16 [NM, ND]   (score B-fragments)
//   VT = Wv @ mem^T + bv    -> bf16 [ND, NM]   (ctx B-fragments contiguous)
// ---------------------------------------------------------------------------
#define KV_ROWS  32
#define SM_PITCH 528   // 512 + 16 bf16 pad (bank spread)

__global__ __launch_bounds__(256) void kvproj_kernel(
    const float* memory, const __bf16* wk_b, const __bf16* wv_b,
    const float* bk, const float* bv,
    __bf16* K_b, __bf16* VT_b, float* newmem) {
  __shared__ __bf16 smem[KV_ROWS * SM_PITCH];
  int mblk = blockIdx.x * KV_ROWS;
  int tid = threadIdx.x;

#if USE_TDM
  __shared__ float sf32[KV_ROWS * ND];
  if (tid == 0) {
    unsigned lds = (unsigned)(unsigned long long)(uintptr_t)(void*)sf32;
    unsigned long long gsrc = (unsigned long long)(uintptr_t)(memory +
                              (size_t)mblk * ND);
    unsigned long long gdst = (unsigned long long)(uintptr_t)(newmem +
                              (size_t)mblk * ND);
    tdm_load_tile(gsrc, lds, KV_ROWS);   // TENSORcnt = 1
    tdm_store_tile(gdst, lds, KV_ROWS);  // in-order after load; TENSORcnt = 2
    __builtin_amdgcn_s_wait_tensorcnt(1);  // load landed; store drains async
  }
  __syncthreads();
  // convert f32 tile -> padded bf16 tile
  for (int it = tid; it < KV_ROWS * 128; it += 256) {
    int r = it >> 7, c4 = it & 127;
    float4 f = reinterpret_cast<const float4*>(sf32 + r * ND)[c4];
    __bf16* d = smem + r * SM_PITCH + c4 * 4;
    d[0] = (__bf16)f.x; d[1] = (__bf16)f.y; d[2] = (__bf16)f.z; d[3] = (__bf16)f.w;
  }
#else
  // fallback: manual stage-in + passthrough copy
  for (int it = tid; it < KV_ROWS * 128; it += 256) {
    int r = it >> 7, c4 = it & 127;
    float4 f = reinterpret_cast<const float4*>(memory + (size_t)(mblk + r) * ND)[c4];
    reinterpret_cast<float4*>(newmem + (size_t)(mblk + r) * ND)[c4] = f;
    __bf16* d = smem + r * SM_PITCH + c4 * 4;
    d[0] = (__bf16)f.x; d[1] = (__bf16)f.y; d[2] = (__bf16)f.z; d[3] = (__bf16)f.w;
  }
#endif
  __syncthreads();

  int lane = tid & 31, wave = tid >> 5;
  int col = lane & 15, seg = lane >> 4;

  // K tiles: 2 mtiles x 32 dtiles. LDS rows feed A-fragments.
  for (int t = wave; t < 64; t += 8) {
    int mt = t >> 5, dt = t & 31;
    float bias = bk[dt * 16 + col];
    v8f acc;
    #pragma unroll
    for (int r = 0; r < 8; r++) acc[r] = bias;
    #pragma unroll
    for (int ks = 0; ks < 16; ks++) {
      v16bf a = load_frag(smem + (mt * 16 + col) * SM_PITCH + ks * 32 + seg * 16);
      v16bf b = load_frag(wk_b + (dt * 16 + col) * ND + ks * 32 + seg * 16);
      acc = WMMA_BF16(a, b, acc);
    }
    #pragma unroll
    for (int r = 0; r < 8; r++) {
      int mg = mblk + mt * 16 + r + 8 * seg;
      K_b[(size_t)mg * ND + dt * 16 + col] = (__bf16)acc[r];
    }
  }
  // VT tiles: 32 dtiles x 2 mtiles. Same LDS layout now feeds B-fragments.
  for (int t = wave; t < 64; t += 8) {
    int dt = t >> 1, mt = t & 1;
    v8f acc;
    #pragma unroll
    for (int r = 0; r < 8; r++) acc[r] = bv[dt * 16 + r + 8 * seg];
    #pragma unroll
    for (int ks = 0; ks < 16; ks++) {
      v16bf a = load_frag(wv_b + (dt * 16 + col) * ND + ks * 32 + seg * 16);
      v16bf b = load_frag(smem + (mt * 16 + col) * SM_PITCH + ks * 32 + seg * 16);
      acc = WMMA_BF16(a, b, acc);
    }
    #pragma unroll
    for (int r = 0; r < 8; r++) {
      int dg = dt * 16 + r + 8 * seg;
      int mg = mblk + mt * 16 + col;
      VT_b[(size_t)dg * NM + mg] = (__bf16)acc[r];
    }
  }
}

// ---------------------------------------------------------------------------
// flash: split-K flash attention. grid (SPLITS, 8 btiles, 8 heads), 8 waves.
// Record per (h,bt,split): 1024 ctx floats + 16 rowmax + 16 rowsum.
// ---------------------------------------------------------------------------
#define P_PITCH 136   // 128 + 8 bf16 pad

__global__ __launch_bounds__(256) void flash_kernel(
    const __bf16* q_b, const __bf16* K_b, const __bf16* VT_b, float* partial) {
  // per-wave 1088 floats = 4352B: first P staging (bf16), then ctx f32
  __shared__ float shbuf[8 * 1088];
  __shared__ float mbuf[8][16], lbuf[8][16], wscale[8][16], mstar[16], lsum[16];

  int split = blockIdx.x, bt = blockIdx.y, h = blockIdx.z;
  int tid = threadIdx.x, lane = tid & 31, wave = tid >> 5;
  int col = lane & 15, seg = lane >> 4;
  int mbase = split * 1024 + wave * 128;

  v16bf qa0 = load_frag(q_b + (bt * 16 + col) * ND + h * HD + seg * 16);
  v16bf qa1 = load_frag(q_b + (bt * 16 + col) * ND + h * HD + 32 + seg * 16);

  v8f s[8];
  #pragma unroll
  for (int t = 0; t < 8; t++) {
    v8f acc;
    #pragma unroll
    for (int r = 0; r < 8; r++) acc[r] = 0.f;
    const __bf16* kp = K_b + (size_t)(mbase + t * 16 + col) * ND + h * HD;
    acc = WMMA_BF16(qa0, load_frag(kp + seg * 16), acc);
    acc = WMMA_BF16(qa1, load_frag(kp + 32 + seg * 16), acc);
    s[t] = acc;
  }

  // rows live on 16-lane halves -> xor masks 1,2,4,8 only
  float rmax[8], rsum[8];
  __bf16* myP = reinterpret_cast<__bf16*>(shbuf + wave * 1088);
  #pragma unroll
  for (int r = 0; r < 8; r++) {
    float mx = s[0][r];
    #pragma unroll
    for (int t = 1; t < 8; t++) mx = fmaxf(mx, s[t][r]);
    #pragma unroll
    for (int off = 1; off < 16; off <<= 1) mx = fmaxf(mx, __shfl_xor(mx, off, 32));
    rmax[r] = mx;
    float sum = 0.f;
    int R = r + 8 * seg;
    #pragma unroll
    for (int t = 0; t < 8; t++) {
      float p = __expf(s[t][r] - mx);
      sum += p;
      myP[R * P_PITCH + t * 16 + col] = (__bf16)p;
    }
    #pragma unroll
    for (int off = 1; off < 16; off <<= 1) sum += __shfl_xor(sum, off, 32);
    rsum[r] = sum;
  }

  // ctx = P[16x128] x V[128x64]
  v8f cacc[4];
  #pragma unroll
  for (int dt = 0; dt < 4; dt++)
    #pragma unroll
    for (int r = 0; r < 8; r++) cacc[dt][r] = 0.f;
  #pragma unroll
  for (int kk = 0; kk < 4; kk++) {
    v16bf a = load_frag(myP + col * P_PITCH + kk * 32 + seg * 16);
    #pragma unroll
    for (int dt = 0; dt < 4; dt++) {
      v16bf b = load_frag(VT_b + (size_t)(h * HD + dt * 16 + col) * NM +
                          mbase + kk * 32 + seg * 16);
      cacc[dt] = WMMA_BF16(a, b, cacc[dt]);
    }
  }
  __syncthreads();   // fence: P reads done before f32 reuse of shbuf

  float* cw = shbuf + wave * 1088;
  #pragma unroll
  for (int dt = 0; dt < 4; dt++)
    #pragma unroll
    for (int r = 0; r < 8; r++)
      cw[(r + 8 * seg) * 64 + dt * 16 + col] = cacc[dt][r];
  if (col == 0) {
    #pragma unroll
    for (int r = 0; r < 8; r++) {
      mbuf[wave][r + 8 * seg] = rmax[r];
      lbuf[wave][r + 8 * seg] = rsum[r];
    }
  }
  __syncthreads();

  if (tid < 16) {
    float mm = mbuf[0][tid];
    for (int w = 1; w < 8; w++) mm = fmaxf(mm, mbuf[w][tid]);
    float ll = 0.f;
    for (int w = 0; w < 8; w++) {
      float sc = __expf(mbuf[w][tid] - mm);
      wscale[w][tid] = sc;
      ll += sc * lbuf[w][tid];
    }
    mstar[tid] = mm;
    lsum[tid] = ll;
  }
  __syncthreads();

  float* rec = partial + (size_t)(((h * 8 + bt) * SPLITS) + split) * 1056;
  for (int e = tid; e < 1024; e += 256) {
    int r = e >> 6, d = e & 63;
    float v = 0.f;
    #pragma unroll
    for (int w = 0; w < 8; w++) v += wscale[w][r] * shbuf[w * 1088 + r * 64 + d];
    rec[e] = v;
  }
  if (tid < 16) { rec[1024 + tid] = mstar[tid]; rec[1040 + tid] = lsum[tid]; }
}

// ---------------------------------------------------------------------------
// combine: merge SPLITS partials per (h, btile) -> normalized ctx, bf16 out
// ---------------------------------------------------------------------------
__global__ __launch_bounds__(256) void combine_kernel(const float* partial,
                                                      __bf16* ctx_b) {
  __shared__ float esc[SPLITS][16];
  __shared__ float Ls[16];
  int hb = blockIdx.x;
  int h = hb >> 3, bt = hb & 7;
  const float* base = partial + (size_t)hb * SPLITS * 1056;
  int tid = threadIdx.x;
  if (tid < 16) {
    float mm = -1e30f;
    for (int s = 0; s < SPLITS; s++) mm = fmaxf(mm, base[s * 1056 + 1024 + tid]);
    float ll = 0.f;
    for (int s = 0; s < SPLITS; s++) {
      float sc = __expf(base[s * 1056 + 1024 + tid] - mm);
      esc[s][tid] = sc;
      ll += sc * base[s * 1056 + 1040 + tid];
    }
    Ls[tid] = ll;
  }
  __syncthreads();
  for (int e = tid; e < 1024; e += 256) {
    int r = e >> 6, d = e & 63;
    float acc = 0.f;
    for (int s = 0; s < SPLITS; s++) acc += esc[s][r] * base[s * 1056 + e];
    ctx_b[(size_t)(bt * 16 + r) * ND + h * HD + d] = (__bf16)(acc / Ls[r]);
  }
}

// ---------------------------------------------------------------------------
// outproj: out = ctx @ Wo^T + bo  (f32 to d_out)
// ---------------------------------------------------------------------------
__global__ __launch_bounds__(256) void outproj_kernel(const __bf16* ctx_b,
                                                      const __bf16* wo_b,
                                                      const float* bo, float* out) {
  int lane = threadIdx.x & 31, wave = threadIdx.x >> 5;
  int tile = blockIdx.x * 8 + wave;
  int bt = tile >> 5, dt = tile & 31;
  int col = lane & 15, seg = lane >> 4;
  float bias = bo[dt * 16 + col];
  v8f acc;
  #pragma unroll
  for (int r = 0; r < 8; r++) acc[r] = bias;
  #pragma unroll
  for (int ks = 0; ks < 16; ks++) {
    v16bf a = load_frag(ctx_b + (bt * 16 + col) * ND + ks * 32 + seg * 16);
    v16bf b = load_frag(wo_b  + (dt * 16 + col) * ND + ks * 32 + seg * 16);
    acc = WMMA_BF16(a, b, acc);
  }
  #pragma unroll
  for (int r = 0; r < 8; r++)
    out[(bt * 16 + r + 8 * seg) * ND + dt * 16 + col] = acc[r];
}

// ---------------------------------------------------------------------------
// Deterministic top-128-smallest usage: histogram on monotone float bits,
// cutoff bin, exact (value,index)-lexicographic rank, scatter.
// ---------------------------------------------------------------------------
__global__ void hist_kernel(const float* usage, unsigned* hist) {
  int i = blockIdx.x * 256 + threadIdx.x;
  if (i < NM) atomicAdd(&hist[__float_as_uint(usage[i]) >> 20], 1u);
}

__global__ void scan_kernel(const unsigned* hist, unsigned* cutoff) {
  if (threadIdx.x == 0 && blockIdx.x == 0) {
    unsigned c = 0, b = 0;
    for (b = 0; b < 4096; b++) { c += hist[b]; if (c >= (unsigned)NB) break; }
    *cutoff = b;
  }
}

__global__ void select_kernel(const float* usage, const float* feat,
                              const float* importance, const unsigned* cutoff,
                              float* newmem, float* newusage) {
  int i = blockIdx.x * 256 + threadIdx.x;
  if (i >= NM) return;
  float ui = usage[i];
  if ((__float_as_uint(ui) >> 20) > *cutoff) return;
  int rank = 0;
  for (int j = 0; j < NM; j++) {
    float uj = usage[j];
    if (uj < ui || (uj == ui && j < i)) {
      if (++rank >= NB) return;    // rank only grows; not in top-128
    }
  }
  const float4* src = reinterpret_cast<const float4*>(feat + (size_t)rank * ND);
  float4* dst = reinterpret_cast<float4*>(newmem + (size_t)i * ND);
  for (int t = 0; t < ND / 4; t++) dst[t] = src[t];
  newusage[i] = importance[rank];
}

// ---------------------------------------------------------------------------
extern "C" void kernel_launch(void* const* d_in, const int* in_sizes, int n_in,
                              void* d_out, int out_size, void* d_ws, size_t ws_size,
                              hipStream_t stream) {
  const float* features   = (const float*)d_in[0];
  const float* importance = (const float*)d_in[1];
  const float* memory     = (const float*)d_in[2];
  const float* usage      = (const float*)d_in[3];
  const float* Wq = (const float*)d_in[4];
  const float* Wk = (const float*)d_in[5];
  const float* Wv = (const float*)d_in[6];
  const float* bq = (const float*)d_in[7];
  const float* bk = (const float*)d_in[8];
  const float* bv = (const float*)d_in[9];
  const float* Wo = (const float*)d_in[10];
  const float* bo = (const float*)d_in[11];

  float* out      = (float*)d_out;                 // [128,512]
  float* newmem   = out + (size_t)NB * ND;         // [65536,512]
  float* newusage = newmem + (size_t)NM * ND;      // [65536]

  // workspace layout (~150 MiB)
  char* ws = (char*)d_ws;
  const size_t KB512 = (size_t)ND * ND * 2;        // 512 KiB per bf16 weight
  __bf16* wq_b   = (__bf16*)(ws + 0 * KB512);
  __bf16* wk_b   = (__bf16*)(ws + 1 * KB512);
  __bf16* wv_b   = (__bf16*)(ws + 2 * KB512);
  __bf16* wo_b   = (__bf16*)(ws + 3 * KB512);
  __bf16* feat_b = (__bf16*)(ws + 4 * KB512);
  __bf16* q_b    = (__bf16*)(ws + 4 * KB512 + 131072);
  __bf16* ctx_b  = (__bf16*)(ws + 4 * KB512 + 2 * 131072);
  unsigned* hist   = (unsigned*)(ws + 4 * KB512 + 3 * 131072);
  unsigned* cutoff = (unsigned*)(ws + 4 * KB512 + 3 * 131072 + 4096 * 4);
  __bf16* K_b    = (__bf16*)(ws + (4u << 20));                         // 64 MiB
  __bf16* VT_b   = (__bf16*)(ws + (4u << 20) + (size_t)NM * ND * 2);   // 64 MiB
  float* partial = (float*)(ws + (4u << 20) + 2 * (size_t)NM * ND * 2);

  prep_kernel<<<1024, 256, 0, stream>>>(Wq, Wk, Wv, Wo, features, usage,
                                        wq_b, wk_b, wv_b, wo_b, feat_b,
                                        newusage, hist);
  qproj_kernel<<<32, 256, 0, stream>>>(feat_b, wq_b, bq, q_b);
  kvproj_kernel<<<NM / KV_ROWS, 256, 0, stream>>>(memory, wk_b, wv_b, bk, bv,
                                                  K_b, VT_b, newmem);
  flash_kernel<<<dim3(SPLITS, 8, 8), 256, 0, stream>>>(q_b, K_b, VT_b, partial);
  combine_kernel<<<64, 256, 0, stream>>>(partial, ctx_b);
  outproj_kernel<<<32, 256, 0, stream>>>(ctx_b, wo_b, bo, out);
  hist_kernel<<<NM / 256, 256, 0, stream>>>(usage, hist);
  scan_kernel<<<1, 64, 0, stream>>>(hist, cutoff);
  select_kernel<<<NM / 256, 256, 0, stream>>>(usage, features, importance,
                                              cutoff, newmem, newusage);
}
